// DenseGCN_36644660969954
// MI455X (gfx1250) — matrix operand order repaired and checked by reading, not compiled, not run on hardware
//
#include <hip/hip_runtime.h>

// ---------------------------------------------------------------------------
// Fused DenseGCN for gfx1250 (MI455X): bf16 WMMA GEMMs with LayerNorm folded
// into pre-transformed weights; whole concat activation lives in LDS.
// ---------------------------------------------------------------------------

typedef __bf16 bf16_t;
typedef __bf16 v16bf __attribute__((ext_vector_type(16)));
typedef __bf16 v8bf  __attribute__((ext_vector_type(8)));
typedef float  v8f   __attribute__((ext_vector_type(8)));

// ---- problem constants -----------------------------------------------------
constexpr int NN    = 17;       // nodes
constexpr int BATCH = 16384;
constexpr int TS    = 2;        // samples per block
constexpr int ROWS  = TS * NN;  // 34 live rows
constexpr int ROWP  = 48;       // padded to 3 WMMA M-tiles
constexpr int MT    = ROWP / 16;
constexpr int XSTR  = 328;      // bf16 elems per activation row (>=320, mult of 8)
constexpr int HSTR  = 72;       // f32 elems per H row

// stage tables: gcn1, layers[0..3], final
constexpr int DIN[6]   = {120, 64, 128, 192, 256, 320};
constexpr int DINP[6]  = {128, 64, 128, 192, 256, 320};  // K padded to mult of 32
constexpr int DOUTR[6] = { 64, 64,  64,  64,  64,   3};
constexpr int DOUTP[6] = { 64, 64,  64,  64,  64,  16};  // N padded to mult of 16
constexpr int COLB[6]  = {  0, 64, 128, 192, 256,   0};  // concat write column

// ---- workspace layout (prep-kernel outputs) --------------------------------
struct WsOff {
  unsigned wt0[6], wt1[6], sv0[6], cv0[6], sv1[6], cv1[6];
  unsigned bia[6], mw[6], dv[6], off[6];
  unsigned total;
};
constexpr unsigned ws_aln(unsigned x) { return (x + 63u) & ~63u; }
constexpr WsOff mkws() {
  WsOff o{};
  unsigned p = 0;
  for (int s = 0; s < 6; ++s) {
    unsigned dinp = (unsigned)DINP[s], dop = (unsigned)DOUTP[s];
    o.wt0[s] = p; p = ws_aln(p + dop * dinp * 2);
    o.wt1[s] = p; p = ws_aln(p + dop * dinp * 2);
    o.sv0[s] = p; p = ws_aln(p + dop * 4);
    o.cv0[s] = p; p = ws_aln(p + dop * 4);
    o.sv1[s] = p; p = ws_aln(p + dop * 4);
    o.cv1[s] = p; p = ws_aln(p + dop * 4);
    o.bia[s] = p; p = ws_aln(p + dop * 4);
    o.mw[s]  = p; p = ws_aln(p + NN * dop * 4);
    o.dv[s]  = p; p = ws_aln(p + NN * 4);
    o.off[s] = p; p = ws_aln(p + NN * NN * 4);
  }
  o.total = p;
  return o;
}
constexpr WsOff WSO = mkws();

__device__ __forceinline__ v16bf cat8(v8bf a, v8bf b) {
  return __builtin_shufflevector(a, b, 0, 1, 2, 3, 4, 5, 6, 7,
                                        8, 9, 10, 11, 12, 13, 14, 15);
}

// ---------------------------------------------------------------------------
// Prep kernel (one per stage): bake LN gamma into transposed bf16 weights,
// compute s = 1^T(g*W), c = b_ln^T W, pad M/bias, symmetrize adjacency.
// ---------------------------------------------------------------------------
template <int S>
__global__ __launch_bounds__(256) void prep_stage(
    const float* __restrict__ W,    const float* __restrict__ M,
    const float* __restrict__ adj2, const float* __restrict__ bvec,
    const float* __restrict__ ln_g, const float* __restrict__ ln_b,
    const float* __restrict__ adj,  char* __restrict__ ws) {
  constexpr int din = DIN[S], dinp = DINP[S], dout = DOUTR[S], doutp = DOUTP[S];
  bf16_t* wt0 = (bf16_t*)(ws + WSO.wt0[S]);
  bf16_t* wt1 = (bf16_t*)(ws + WSO.wt1[S]);
  float* sv0 = (float*)(ws + WSO.sv0[S]);
  float* cv0 = (float*)(ws + WSO.cv0[S]);
  float* sv1 = (float*)(ws + WSO.sv1[S]);
  float* cv1 = (float*)(ws + WSO.cv1[S]);
  float* bia = (float*)(ws + WSO.bia[S]);
  float* mw  = (float*)(ws + WSO.mw[S]);
  float* dv  = (float*)(ws + WSO.dv[S]);
  float* off = (float*)(ws + WSO.off[S]);
  const int tid = threadIdx.x;

  for (int idx = tid; idx < 2 * doutp; idx += 256) {
    int wm = idx / doutp, f = idx - wm * doutp;
    const float* Wm = W + (long)wm * din * dout;
    bf16_t* wt = wm ? wt1 : wt0;
    float ss = 0.f, cc = 0.f;
    for (int k = 0; k < dinp; ++k) {
      float val = 0.f;
      if (f < dout && k < din) {
        float wkf = Wm[k * dout + f];
        float g = ln_g ? ln_g[k] : 1.f;
        float bb = ln_b ? ln_b[k] : 0.f;
        val = wkf * g;
        ss += val;
        cc += bb * wkf;
      }
      wt[(long)f * dinp + k] = (bf16_t)val;
    }
    if (wm) { sv1[f] = ss; cv1[f] = cc; }
    else    { sv0[f] = ss; cv0[f] = cc; }
  }
  for (int idx = tid; idx < doutp; idx += 256)
    bia[idx] = (idx < dout) ? bvec[idx] : 0.f;
  for (int idx = tid; idx < NN * doutp; idx += 256) {
    int n = idx / doutp, f = idx - n * doutp;
    mw[idx] = (f < dout) ? M[n * dout + f] : 0.f;
  }
  for (int idx = tid; idx < NN * NN; idx += 256) {
    int i2 = idx / NN, j = idx - i2 * NN;
    float A = 0.5f * ((adj[i2 * NN + j] + adj2[i2 * NN + j]) +
                      (adj[j * NN + i2] + adj2[j * NN + i2]));
    if (i2 == j) { dv[i2] = A; off[idx] = 0.f; }
    else         { off[idx] = A; }
  }
}

// ---------------------------------------------------------------------------
// Main fused kernel: 2 samples (34 rows) per block, 8 wave32s, 6 stages.
// ---------------------------------------------------------------------------
__global__ __launch_bounds__(256, 1) void gcn_fused(
    const float* __restrict__ xg_all, float* __restrict__ outg,
    const char* __restrict__ ws) {
  __shared__ bf16_t xs[ROWP * XSTR];      // concat activation (bf16, WMMA A)
  __shared__ float  H0[ROWP * HSTR];      // raw x@Wg0 -> then d*M*h0 + bias
  __shared__ float  H1[ROWP * HSTR];      // raw x@Wg1 -> then M*h1
  __shared__ float  smu[ROWS], sis[ROWS];
  __shared__ float  ps[ROWS][4], pq[ROWS][4];
  __shared__ float  soff[NN * NN], sdv[NN];

  const int tid  = threadIdx.x;
  const int lane = tid & 31;
  const int wave = tid >> 5;
  const long blk = blockIdx.x;

  // zero activation buffer (pad rows/cols must be exact zeros for WMMA)
  for (int i = tid; i < ROWP * XSTR; i += 256) xs[i] = (bf16_t)0.f;
  __syncthreads();
  const float* xg = xg_all + blk * (long)(TS * NN * 120);
  for (int i = tid; i < TS * NN * 120; i += 256) {
    int row = i / 120, col = i - row * 120;
    xs[row * XSTR + col] = (bf16_t)xg[i];
  }
  __syncthreads();

#pragma unroll
  for (int s = 0; s < 6; ++s) {
    const int din = DIN[s], dinp = DINP[s], doutp = DOUTP[s];
    const int nt = doutp >> 4;
    const bf16_t* wt0 = (const bf16_t*)(ws + WSO.wt0[s]);
    const bf16_t* wt1 = (const bf16_t*)(ws + WSO.wt1[s]);
    const float* sv0 = (const float*)(ws + WSO.sv0[s]);
    const float* cv0 = (const float*)(ws + WSO.cv0[s]);
    const float* sv1 = (const float*)(ws + WSO.sv1[s]);
    const float* cv1 = (const float*)(ws + WSO.cv1[s]);
    const float* bia = (const float*)(ws + WSO.bia[s]);
    const float* mw  = (const float*)(ws + WSO.mw[s]);
    const float* dvp = (const float*)(ws + WSO.dv[s]);
    const float* ofp = (const float*)(ws + WSO.off[s]);

    // cache adjacency pieces in LDS (visible after the post-GEMM barrier)
    for (int i = tid; i < NN * NN; i += 256) soff[i] = ofp[i];
    if (tid < NN) sdv[tid] = dvp[tid];

    // ---- LayerNorm stats (padding cols are zero, so sum over dinp is exact)
    if (s < 5) {
      if (tid < ROWS * 4) {
        int row = tid >> 2, part = tid & 3;
        int cpt = dinp >> 2;
        const bf16_t* xr = xs + row * XSTR + part * cpt;
        float sm = 0.f, sq = 0.f;
        for (int k = 0; k < cpt; ++k) { float v = (float)xr[k]; sm += v; sq += v * v; }
        ps[row][part] = sm; pq[row][part] = sq;
      }
      __syncthreads();
      if (tid < ROWS) {
        float sm = ps[tid][0] + ps[tid][1] + ps[tid][2] + ps[tid][3];
        float sq = pq[tid][0] + pq[tid][1] + pq[tid][2] + pq[tid][3];
        float mu = sm / (float)din;
        float var = sq / (float)din - mu * mu;
        smu[tid] = mu;
        sis[tid] = rsqrtf(var + 1e-5f);
      }
    } else {
      if (tid < ROWS) { smu[tid] = 0.f; sis[tid] = 1.f; }
    }

    // ---- WMMA GEMMs: jobs = (Mtile, Ntile, {W0,W1}) striped over 8 waves
    const int njobs = MT * nt * 2;
    const int mrow = lane & 15;
    const int kbA = (lane >> 4) << 3;   // A: K chunks {0..7,16..23}/{8..15,24..31}
    const int kbB = (lane >> 4) << 3;   // B: K chunks {0..7,16..23}/{8..15,24..31}
    for (int j = wave; j < njobs; j += 8) {
      const int m = j / (nt * 2);
      const int rem = j - m * (nt * 2);
      const int n0 = (rem >> 1) << 4;
      const int wm = rem & 1;
      const bf16_t* wt = wm ? wt1 : wt0;
      const bf16_t* xrow = xs + (m * 16 + mrow) * XSTR;
      const bf16_t* brow = wt + (long)(n0 + mrow) * dinp;
      v8f acc = {0.f, 0.f, 0.f, 0.f, 0.f, 0.f, 0.f, 0.f};
      for (int kk = 0; kk < dinp; kk += 32) {
        v8bf a0 = *(const v8bf*)(xrow + kk + kbA);
        v8bf a1 = *(const v8bf*)(xrow + kk + kbA + 16);
        v8bf b0 = *(const v8bf*)(brow + kk + kbB);
        v8bf b1 = *(const v8bf*)(brow + kk + kbB + 16);
        acc = __builtin_amdgcn_wmma_f32_16x16x32_bf16(
            false, cat8(a0, a1), false, cat8(b0, b1), (short)0, acc, false, false);
      }
      float* Hp = wm ? H1 : H0;
      const int rbase = m * 16 + ((lane >> 4) << 3);
      const int cc = n0 + (lane & 15);
#pragma unroll
      for (int r = 0; r < 8; ++r) Hp[(rbase + r) * HSTR + cc] = acc[r];
    }
    __syncthreads();

    // ---- epilogue 1: apply folded LN, M-modulation, diagonal term + bias
    for (int i = tid; i < ROWS * doutp; i += 256) {
      int row = i / doutp, f = i - row * doutp;
      int n = (row < NN) ? row : row - NN;
      float isr = sis[row], mis = smu[row] * isr;
      float h0 = H0[row * HSTR + f] * isr - mis * sv0[f] + cv0[f];
      float h1 = H1[row * HSTR + f] * isr - mis * sv1[f] + cv1[f];
      float mnf = mw[n * doutp + f];
      H1[row * HSTR + f] = mnf * h1;
      H0[row * HSTR + f] = sdv[n] * mnf * h0 + bia[f];
    }
    __syncthreads();

    // ---- epilogue 2: 17x17 off-diagonal node mix, GELU, concat / store
    for (int i = tid; i < ROWS * doutp; i += 256) {
      int row = i / doutp, f = i - row * doutp;
      int n = (row < NN) ? row : row - NN;
      int sb = (row < NN) ? 0 : NN;
      float acc2 = H0[row * HSTR + f];
#pragma unroll
      for (int jj = 0; jj < NN; ++jj)
        acc2 += soff[n * NN + jj] * H1[(sb + jj) * HSTR + f];
      if (s < 5) {
        float ge = 0.5f * acc2 * (1.f + erff(acc2 * 0.70710678118f));
        xs[row * XSTR + COLB[s] + f] = (bf16_t)ge;
      } else if (f < 3) {
        outg[(blk * TS + ((row < NN) ? 0 : 1)) * 51 + n * 3 + f] = acc2;
      }
    }
    __syncthreads();
  }
}

// ---------------------------------------------------------------------------
// Host launcher.  Input leaf order assumption: top-level dict insertion order
// (x_gcn, adj, params), params flattened jax-style (dict keys sorted):
//  2: final.M  3: final.W  4: final.adj2  5: final.b
//  6: gcn1.M   7: gcn1.W   8: gcn1.adj2   9: gcn1.b
//  10+6i: layers[i] {M, W, adj2, b, ln_b, ln_g}
//  34: ln0_b   35: ln0_g
// ---------------------------------------------------------------------------
extern "C" void kernel_launch(void* const* d_in, const int* in_sizes, int n_in,
                              void* d_out, int out_size, void* d_ws, size_t ws_size,
                              hipStream_t stream) {
  (void)in_sizes; (void)n_in; (void)out_size; (void)ws_size;
  const float* x   = (const float*)d_in[0];
  const float* adj = (const float*)d_in[1];
  char* ws = (char*)d_ws;
  auto F = [&](int i) { return (const float*)d_in[i]; };

  // stage 0: gcn1 (LN0)
  prep_stage<0><<<1, 256, 0, stream>>>(F(7), F(6), F(8), F(9), F(35), F(34), adj, ws);
  // stages 1..4: layers[0..3]
  prep_stage<1><<<1, 256, 0, stream>>>(F(11), F(10), F(12), F(13), F(15), F(14), adj, ws);
  prep_stage<2><<<1, 256, 0, stream>>>(F(17), F(16), F(18), F(19), F(21), F(20), adj, ws);
  prep_stage<3><<<1, 256, 0, stream>>>(F(23), F(22), F(24), F(25), F(27), F(26), adj, ws);
  prep_stage<4><<<1, 256, 0, stream>>>(F(29), F(28), F(30), F(31), F(33), F(32), adj, ws);
  // stage 5: final (no LN)
  prep_stage<5><<<1, 256, 0, stream>>>(F(3), F(2), F(4), F(5), nullptr, nullptr, adj, ws);

  gcn_fused<<<dim3(BATCH / TS), dim3(256), 0, stream>>>(x, (float*)d_out, ws);
}
